// Attention_86629490360820
// MI455X (gfx1250) — matrix operand order repaired and checked by reading, not compiled, not run on hardware
//
#include <hip/hip_runtime.h>

typedef __bf16 bf16;
typedef __attribute__((ext_vector_type(16))) __bf16 v16bf;
typedef __attribute__((ext_vector_type(8)))  float  v8f;
typedef __attribute__((ext_vector_type(4)))  uint32_t u32x4;
typedef __attribute__((ext_vector_type(8)))  uint32_t u32x8;

#define DEV_INLINE __device__ __forceinline__

constexpr int Bb   = 2;
constexpr int Nn   = 2048;
constexpr int HIDc = 1024;
constexpr int Hh   = 16;
constexpr int DHc  = 64;
constexpr int E3   = 3 * Hh * DHc;   // 3072
constexpr int Mrows = Bb * Nn;       // 4096

// ---- WMMA fragment helpers (gfx1250 wave32 layouts, CDNA5 ISA) ----
DEV_INLINE v16bf load_a_frag(const bf16* base, int ld, int lane) {
  int row = lane & 15, g = lane >> 4;
  const bf16* p0 = base + row * ld + 8 * g;
  const bf16* p1 = base + row * ld + 16 + 8 * g;
  union { v16bf v; uint4 u[2]; } t;
  t.u[0] = *(const uint4*)p0;
  t.u[1] = *(const uint4*)p1;
  return t.v;
}

DEV_INLINE v16bf load_bt_frag(const bf16* base, int ld, int lane) {
  int col = lane & 15, g = lane >> 4;
  const bf16* p0 = base + col * ld + 8 * g;
  const bf16* p1 = base + col * ld + 16 + 8 * g;
  union { v16bf v; uint4 u[2]; } t;
  t.u[0] = *(const uint4*)p0;
  t.u[1] = *(const uint4*)p1;
  return t.v;
}

DEV_INLINE v8f wmma_bf16(v16bf a, v16bf b, v8f c) {
  return __builtin_amdgcn_wmma_f32_16x16x32_bf16(
      false, a, false, b, (short)0, c, false, false);
}

// Async global->LDS copy, 16 bytes per lane (ASYNCcnt-tracked).
DEV_INLINE void async_copy_b128(void* lds, const void* gaddr) {
  uint32_t lds_off = (uint32_t)(uintptr_t)lds;
  asm volatile("global_load_async_to_lds_b128 %0, %1, off"
               :: "v"(lds_off), "v"(gaddr) : "memory");
}
DEV_INLINE void wait_async0() {
  asm volatile("s_wait_asynccnt 0" ::: "memory");
}

// Tensor Data Mover: 2D bf16 tile load global->LDS (TENSORcnt-tracked).
// D# per CDNA5 ISA: group0 = count/lds/global/type,
// group1 = data_size + tensor dims/strides + tile dims (bit-packed).
DEV_INLINE void tdm_load_2d_bf16(void* lds, const void* gptr,
                                 uint32_t tile_w, uint32_t tile_h,
                                 uint32_t row_stride_elems) {
  uint64_t ga = (uint64_t)(uintptr_t)gptr;
  uint32_t la = (uint32_t)(uintptr_t)lds;       // LDS byte offset
  u32x4 g0;
  g0[0] = 1u;                                            // count=1, user D#
  g0[1] = la;                                            // lds_addr[31:0]
  g0[2] = (uint32_t)ga;                                  // global_addr[31:0]
  g0[3] = ((uint32_t)(ga >> 32) & 0x01FFFFFFu)           // global_addr[56:32]
        | (2u << 30);                                    // type=2 ("image")
  u32x8 g1;
  g1[0] = (1u << 16);                                    // data_size=1 (2B)
  g1[1] = (tile_w & 0xFFFFu) << 16;                      // tensor_dim0[15:0]@48
  g1[2] = (tile_w >> 16) | ((tile_h & 0xFFFFu) << 16);   // dim0 hi | dim1 lo
  g1[3] = (tile_h >> 16) | (tile_w << 16);               // dim1 hi | tile_dim0
  g1[4] = (tile_h & 0xFFFFu);                            // tile_dim1, tile_dim2=0
  g1[5] = row_stride_elems;                              // dim0_stride[31:0]
  g1[6] = 0u;                                            // stride hi bits
  g1[7] = 0u;
  asm volatile("tensor_load_to_lds %0, %1"
               :: "s"(g0), "s"(g1) : "memory");
}

DEV_INLINE uint2 pack4_bf16(float4 f) {
  union { bf16 h[4]; uint2 u; } t;
  t.h[0] = (bf16)f.x; t.h[1] = (bf16)f.y; t.h[2] = (bf16)f.z; t.h[3] = (bf16)f.w;
  return t.u;
}

// =====================================================================
// Kernel 1: qkv = x @ Wqkv^T, fused RoPE on q,k. Writes bf16:
//   q,k -> [B,H,N,DH] (roped), v -> [B,H,DH,N] (transposed).
// =====================================================================
__global__ __launch_bounds__(256) void qkv_rope_kernel(
    const float* __restrict__ x, const float* __restrict__ rope,
    const float* __restrict__ Wqkv,
    bf16* __restrict__ qo, bf16* __restrict__ ko, bf16* __restrict__ vTo)
{
  __shared__ bf16 As[64][64];   // 8 KB
  __shared__ bf16 Bs[32][64];   // 4 KB
  const int tid = threadIdx.x;
  const int lane = tid & 31, w = tid >> 5;
  const int wr = w >> 1, wc = w & 1;
  const int m0 = blockIdx.y * 64;
  const int e0 = blockIdx.x * 32;

  v8f acc = {};
  for (int kk = 0; kk < HIDc; kk += 64) {
    __syncthreads();
#pragma unroll
    for (int i = 0; i < 4; ++i) {
      int s = (i << 8) + tid;
      int r = s >> 4, c = (s & 15) * 4;
      float4 f = *(const float4*)&x[(size_t)(m0 + r) * HIDc + kk + c];
      *(uint2*)&As[r][c] = pack4_bf16(f);
    }
#pragma unroll
    for (int i = 0; i < 2; ++i) {
      int s = (i << 8) + tid;
      int r = s >> 4, c = (s & 15) * 4;
      float4 f = *(const float4*)&Wqkv[(size_t)(e0 + r) * HIDc + kk + c];
      *(uint2*)&Bs[r][c] = pack4_bf16(f);
    }
    if (kk + 64 < HIDc)
      __builtin_prefetch(&x[(size_t)(m0 + (tid >> 2)) * HIDc + kk + 64], 0, 0);
    __syncthreads();
    v16bf a0 = load_a_frag(&As[wr * 16][0], 64, lane);
    v16bf a1 = load_a_frag(&As[wr * 16][0] + 32, 64, lane);
    v16bf b0 = load_bt_frag(&Bs[wc * 16][0], 64, lane);
    v16bf b1 = load_bt_frag(&Bs[wc * 16][0] + 32, 64, lane);
    acc = wmma_bf16(a0, b0, acc);
    acc = wmma_bf16(a1, b1, acc);
  }

  const int g = lane >> 4, n = lane & 15;
  const int e = e0 + wc * 16 + n;
  const int sel = e >> 10;        // 0=q 1=k 2=v (uniform per wave)
  const int rem = e & 1023;
  const int h = rem >> 6, dh = rem & 63;
  const int mb = m0 + wr * 16 + 8 * g;
  const int bidx = mb >> 11, tokb = mb & (Nn - 1);

  if (sel == 2) {
    union { bf16 h8[8]; uint4 u; } t;
#pragma unroll
    for (int r = 0; r < 8; ++r) t.h8[r] = (bf16)acc[r];
    *(uint4*)&vTo[(size_t)((bidx * Hh + h) * DHc + dh) * Nn + tokb] = t.u;
  } else {
    bf16* dst = (sel == 0) ? qo : ko;
    const float* rb = rope + (size_t)((bidx << 11) + tokb) * DHc + (dh & ~1);
#pragma unroll
    for (int r = 0; r < 8; ++r) {
      float val = acc[r];
      float other = __shfl_xor(val, 1);
      if ((dh & 1) == 0) {
        float2 csn = *(const float2*)(rb + (size_t)r * DHc);
        float r0 = val * csn.x - other * csn.y;
        float r1 = other * csn.x + val * csn.y;
        union { bf16 h2[2]; uint32_t u; } t;
        t.h2[0] = (bf16)r0; t.h2[1] = (bf16)r1;
        *(uint32_t*)&dst[(size_t)((bidx * Hh + h) * Nn + tokb + r) * DHc + dh] = t.u;
      }
    }
  }
}

// =====================================================================
// Kernel 2: flash attention; K/V tiles staged by the Tensor Data Mover
// (one tensor_load_to_lds per tile, double-buffered, TENSORcnt sync).
// =====================================================================
__global__ __launch_bounds__(256) void attn_kernel(
    const bf16* __restrict__ q, const bf16* __restrict__ k,
    const bf16* __restrict__ vT, bf16* __restrict__ ao)
{
  __shared__ bf16 Ks[2][32][64];    // [j][dh]  (B^T for QK^T)   8 KB
  __shared__ bf16 Vs[2][64][32];    // [dh][j]  (B^T for P.V)    8 KB
  __shared__ bf16 Ps[8][16][32];    // per-wave P transpose      8 KB

  const int tid = threadIdx.x, lane = tid & 31, w = tid >> 5;
  const int bh = blockIdx.y, b = bh >> 4, h = bh & 15;
  const int q0 = blockIdx.x * 128 + w * 16;

  const bf16* qbase = q + (size_t)((b * Hh + h) * Nn + q0) * DHc;
  v16bf a0 = load_a_frag(qbase, DHc, lane);        // dh 0..31
  v16bf a1 = load_a_frag(qbase + 32, DHc, lane);   // dh 32..63

  const bf16* kbase = k + (size_t)(b * Hh + h) * Nn * DHc;
  const bf16* vbase = vT + (size_t)(b * Hh + h) * DHc * Nn;
  const float scale = 0.125f;

  float mrow[8], lrow[8];
  v8f o0 = {}, o1 = {}, o2 = {}, o3 = {};
#pragma unroll
  for (int r = 0; r < 8; ++r) { mrow[r] = -1e30f; lrow[r] = 0.f; }

  const int g = lane >> 4, n = lane & 15;

  // TDM staging: wave 0 moves the K tile, wave 1 the V^T tile.
  auto stage_kv = [&](int j0, int buf) {
    if (w == 0)
      tdm_load_2d_bf16(&Ks[buf][0][0], kbase + (size_t)j0 * DHc,
                       /*w=*/64, /*h=*/32, /*stride=*/DHc);
    else if (w == 1)
      tdm_load_2d_bf16(&Vs[buf][0][0], vbase + j0,
                       /*w=*/32, /*h=*/64, /*stride=*/Nn);
  };

  stage_kv(0, 0);
  __builtin_amdgcn_s_wait_tensorcnt(0);
  __syncthreads();

  constexpr int ITERS = Nn / 32;
  for (int it = 0; it < ITERS; ++it) {
    const int buf = it & 1;
    if (it + 1 < ITERS) stage_kv((it + 1) * 32, buf ^ 1);   // prefetch next

    v16bf bk0  = load_bt_frag(&Ks[buf][0][0], 64, lane);
    v16bf bk0b = load_bt_frag(&Ks[buf][0][0] + 32, 64, lane);
    v16bf bk1  = load_bt_frag(&Ks[buf][16][0], 64, lane);
    v16bf bk1b = load_bt_frag(&Ks[buf][16][0] + 32, 64, lane);
    v8f s0 = {}, s1 = {};
    s0 = wmma_bf16(a0, bk0, s0);  s0 = wmma_bf16(a1, bk0b, s0);
    s1 = wmma_bf16(a0, bk1, s1);  s1 = wmma_bf16(a1, bk1b, s1);

#pragma unroll
    for (int r = 0; r < 8; ++r) {
      float x0 = s0[r] * scale, x1 = s1[r] * scale;
      float mx = fmaxf(x0, x1);
#pragma unroll
      for (int off = 1; off < 16; off <<= 1)
        mx = fmaxf(mx, __shfl_xor(mx, off));
      float mnew  = fmaxf(mrow[r], mx);
      float alpha = __expf(mrow[r] - mnew);
      float p0 = __expf(x0 - mnew), p1 = __expf(x1 - mnew);
      float sum = p0 + p1;
#pragma unroll
      for (int off = 1; off < 16; off <<= 1)
        sum += __shfl_xor(sum, off);
      lrow[r] = lrow[r] * alpha + sum;
      mrow[r] = mnew;
      o0[r] *= alpha; o1[r] *= alpha; o2[r] *= alpha; o3[r] *= alpha;
      int M = 8 * g + r;
      Ps[w][M][n]      = (bf16)p0;
      Ps[w][M][16 + n] = (bf16)p1;
    }
    __syncthreads();

    v16bf pa  = load_a_frag(&Ps[w][0][0], 32, lane);
    v16bf bv0 = load_bt_frag(&Vs[buf][0][0], 32, lane);
    v16bf bv1 = load_bt_frag(&Vs[buf][16][0], 32, lane);
    v16bf bv2 = load_bt_frag(&Vs[buf][32][0], 32, lane);
    v16bf bv3 = load_bt_frag(&Vs[buf][48][0], 32, lane);
    o0 = wmma_bf16(pa, bv0, o0);
    o1 = wmma_bf16(pa, bv1, o1);
    o2 = wmma_bf16(pa, bv2, o2);
    o3 = wmma_bf16(pa, bv3, o3);

    __builtin_amdgcn_s_wait_tensorcnt(0);  // issuing waves drain TDM
    __syncthreads();                       // everyone sees the new tiles
  }

#pragma unroll
  for (int r = 0; r < 8; ++r) {
    int tok = q0 + 8 * g + r;
    float inv = (lrow[r] > 0.f) ? 1.f / lrow[r] : 0.f;
    bf16* dst = ao + (size_t)(b * Nn + tok) * HIDc + h * DHc;
    dst[n]      = (bf16)(o0[r] * inv);
    dst[16 + n] = (bf16)(o1[r] * inv);
    dst[32 + n] = (bf16)(o2[r] * inv);
    dst[48 + n] = (bf16)(o3[r] * inv);
  }
}

// =====================================================================
// Kernel 3: out = attn_out @ Wout^T, f32 result.
// A tile staged with per-lane async b128 copies; B converted f32->bf16.
// =====================================================================
__global__ __launch_bounds__(256) void outproj_kernel(
    const bf16* __restrict__ ao, const float* __restrict__ Wout,
    float* __restrict__ out)
{
  __shared__ bf16 As[64][64];   // 8 KB
  __shared__ bf16 Bs[32][64];   // 4 KB
  const int tid = threadIdx.x, lane = tid & 31, w = tid >> 5;
  const int wr = w >> 1, wc = w & 1;
  const int m0 = blockIdx.y * 64, d0 = blockIdx.x * 32;

  v8f acc = {};
  for (int kk = 0; kk < HIDc; kk += 64) {
    __syncthreads();
#pragma unroll
    for (int i = 0; i < 2; ++i) {
      int s = (i << 8) + tid;
      int r = s >> 3, c = (s & 7) * 8;
      async_copy_b128(&As[r][c], ao + (size_t)(m0 + r) * HIDc + kk + c);
    }
#pragma unroll
    for (int i = 0; i < 2; ++i) {
      int s = (i << 8) + tid;
      int r = s >> 4, c = (s & 15) * 4;
      float4 f = *(const float4*)&Wout[(size_t)(d0 + r) * HIDc + kk + c];
      *(uint2*)&Bs[r][c] = pack4_bf16(f);
    }
    if (kk + 64 < HIDc)
      __builtin_prefetch(&Wout[(size_t)(d0 + (tid >> 3)) * HIDc + kk + 64], 0, 0);
    wait_async0();
    __syncthreads();
    v16bf a0 = load_a_frag(&As[wr * 16][0], 64, lane);
    v16bf a1 = load_a_frag(&As[wr * 16][0] + 32, 64, lane);
    v16bf b0 = load_bt_frag(&Bs[wc * 16][0], 64, lane);
    v16bf b1 = load_bt_frag(&Bs[wc * 16][0] + 32, 64, lane);
    acc = wmma_bf16(a0, b0, acc);
    acc = wmma_bf16(a1, b1, acc);
  }
  const int g = lane >> 4, n = lane & 15;
#pragma unroll
  for (int r = 0; r < 8; ++r) {
    int m = m0 + wr * 16 + 8 * g + r;
    out[(size_t)m * HIDc + d0 + wc * 16 + n] = acc[r];
  }
}

// =====================================================================
extern "C" void kernel_launch(void* const* d_in, const int* in_sizes, int n_in,
                              void* d_out, int out_size, void* d_ws, size_t ws_size,
                              hipStream_t stream) {
  const float* x    = (const float*)d_in[0];
  const float* rope = (const float*)d_in[1];
  const float* Wqkv = (const float*)d_in[2];
  const float* Wout = (const float*)d_in[3];
  float* out = (float*)d_out;

  const size_t QSZ = (size_t)Bb * Hh * Nn * DHc;  // 4M elems (8 MB bf16)
  bf16* qws  = (bf16*)d_ws;
  bf16* kws  = qws + QSZ;
  bf16* vTws = kws + QSZ;
  bf16* aows = vTws + QSZ;   // total 32 MB of workspace

  qkv_rope_kernel<<<dim3(E3 / 32, Mrows / 64), 256, 0, stream>>>(
      x, rope, Wqkv, qws, kws, vTws);
  attn_kernel<<<dim3(Nn / 128, Bb * Hh), 256, 0, stream>>>(
      qws, kws, vTws, aows);
  outproj_kernel<<<dim3(HIDc / 32, Mrows / 64), 256, 0, stream>>>(
      aows, Wout, out);
}
